// MoELookforward_90168543412579
// MI455X (gfx1250) — matrix operand order repaired and checked by reading, not compiled, not run on hardware
//
#include <hip/hip_runtime.h>
#include <cstdint>
#include <cstddef>

// ---------------------------------------------------------------------------
// MoE-Lookforward for MI455X (gfx1250, wave32, WMMA bf16 + async LDS copies).
// B=2, L=1024, h=768, ff=3072, E=8, topk=2, N=2048 tokens.
// Expert block: 64 tokens x full H, 16 waves, ff-chunks split across 4 blocks,
// async weight staging software-pipelined against the two GEMM phases.
// ---------------------------------------------------------------------------

#define NTOK   2048
#define H      768
#define FF     3072
#define NEXP   8
#define TOPK   2
#define TILE_M 64          // tokens per expert-tile / block
#define NTHR   512         // 16 waves
#define CH     64          // ff chunk width
#define NCH    (FF / CH)   // 48 chunks
#define KT1    (H / 32)    // 24 k-tiles for GEMM1
#define CT1    (CH / 16)   // 4 col-tiles of h1
#define KT2    (CH / 32)   // 2 k-tiles for GEMM2
#define CT2    (H / 16)    // 48 col-tiles of out
#define FRAG_CHUNK (96 * 512)   // ush per prepped chunk block (96 frags x 512)

typedef __attribute__((ext_vector_type(16))) __bf16        v16bf;
typedef __attribute__((ext_vector_type(8)))  float         v8f;
typedef __attribute__((ext_vector_type(4)))  unsigned int  u32x4;
typedef __attribute__((ext_vector_type(2)))  unsigned int  u32x2;

union Frag {
  v16bf v;
  u32x4 q[2];
};

static __device__ __forceinline__ unsigned short f2bf(float f) {
  unsigned u = __builtin_bit_cast(unsigned, f);
  u += 0x7FFFu + ((u >> 16) & 1u);          // round-to-nearest-even
  return (unsigned short)(u >> 16);
}

static __device__ __forceinline__ void ldFrag(Frag& f, const unsigned short* p) {
  const u32x4* q = (const u32x4*)p;         // 2 x ds_load_b128 per lane
  f.q[0] = q[0];
  f.q[1] = q[1];
}

static __device__ __forceinline__ v8f wmma_bf16(const Frag& a, const Frag& b, v8f c) {
  return __builtin_amdgcn_wmma_f32_16x16x32_bf16(false, a.v, false, b.v,
                                                 (short)0, c, false, false);
}

// Async 16-byte copy global -> LDS (per lane), tracked by ASYNCcnt.
// Low 32 bits of a generic pointer to LDS are the DS byte address.
static __device__ __forceinline__ void async16(const unsigned short* g,
                                               unsigned short* l) {
  unsigned lds = (unsigned)(uintptr_t)l;
  asm volatile("global_load_async_to_lds_b128 %0, %1, off"
               :: "v"(lds), "v"(g) : "memory");
}
static __device__ __forceinline__ void wait_async0() {
  asm volatile("s_wait_asynccnt 0x0" ::: "memory");
}

// A-fragment element mapping (16-bit A, 16x32): col kk (0..31) -> (half, elem)
static __device__ __forceinline__ void a_map(int kk, int& half, int& elem) {
  half = (kk >> 3) & 1;
  elem = (kk & 7) | ((kk >> 4) << 3);
}
// B-fragment element mapping (16-bit B, 32x16): row kk (0..31) -> (half, elem)
static __device__ __forceinline__ void b_map(int kk, int& half, int& elem) {
  half = (kk >> 3) & 1;
  elem = (kk & 7) | ((kk >> 4) << 3);
}

// ---------------------------------------------------------------------------
// Kernel 0: zero a float buffer (as float4) + workspace header
// ---------------------------------------------------------------------------
__global__ void __launch_bounds__(256) moe_init(float* __restrict__ buf, int n4,
                                                int* __restrict__ cnt,
                                                float* __restrict__ Psum,
                                                float* __restrict__ klsum) {
  int gid = blockIdx.x * blockDim.x + threadIdx.x;
  float4 z; z.x = 0.f; z.y = 0.f; z.z = 0.f; z.w = 0.f;
  float4* o4 = (float4*)buf;
  for (int i = gid; i < n4; i += gridDim.x * blockDim.x) o4[i] = z;
  if (gid < NEXP) { cnt[gid] = 0; Psum[gid] = 0.0f; }
  if (gid == 0) *klsum = 0.0f;
}

// ---------------------------------------------------------------------------
// Kernel 1: weight prep — f32 -> bf16 in WMMA B-fragment, chunk-contiguous
// layout so expert blocks can async-memcpy straight to LDS. grid=(NCH, NEXP)
// ---------------------------------------------------------------------------
__global__ void __launch_bounds__(256) moe_prep(
    const float* __restrict__ Wfc,     // [E][H][FF]
    const float* __restrict__ Wpj,     // [E][FF][H]
    unsigned short* __restrict__ wfcp,
    unsigned short* __restrict__ wpjp) {
  const int ch = blockIdx.x;
  const int e  = blockIdx.y;
  const int tid = threadIdx.x;
  const int ffc0 = ch * CH;
  const float* WfcE = Wfc + (size_t)e * H * FF;
  const float* WpjE = Wpj + (size_t)e * FF * H;
  unsigned short* dfc = wfcp + (size_t)(e * NCH + ch) * FRAG_CHUNK;
  unsigned short* dpj = wpjp + (size_t)(e * NCH + ch) * FRAG_CHUNK;

  for (int i = tid; i < H * CH / 4; i += 256) {
    const int r  = i / (CH / 4);
    const int c4 = (i - r * (CH / 4)) * 4;
    const float4 v = *(const float4*)&WfcE[(size_t)r * FF + ffc0 + c4];
    const int t = r >> 5, kk = r & 31;
    int half, elem; b_map(kk, half, elem);
    const int c = c4 >> 4, n = c4 & 15;
    unsigned short* base = dfc + (((t * CT1 + c) << 9) + elem);
    base[((half << 4) + n + 0) * 16] = f2bf(v.x);
    base[((half << 4) + n + 1) * 16] = f2bf(v.y);
    base[((half << 4) + n + 2) * 16] = f2bf(v.z);
    base[((half << 4) + n + 3) * 16] = f2bf(v.w);
  }
  for (int i = tid; i < CH * H / 4; i += 256) {
    const int r  = i / (H / 4);
    const int c4 = (i - r * (H / 4)) * 4;
    const float4 v = *(const float4*)&WpjE[(size_t)(ffc0 + r) * H + c4];
    const int t = r >> 5, kk = r & 31;
    int half, elem; b_map(kk, half, elem);
    const int c = c4 >> 4, n = c4 & 15;
    unsigned short* base = dpj + (((t * CT2 + c) << 9) + elem);
    base[((half << 4) + n + 0) * 16] = f2bf(v.x);
    base[((half << 4) + n + 1) * 16] = f2bf(v.y);
    base[((half << 4) + n + 2) * 16] = f2bf(v.z);
    base[((half << 4) + n + 3) * 16] = f2bf(v.w);
  }
}

// ---------------------------------------------------------------------------
// Kernel 2: router (f32) — logits, softmax, top2, KL, load-balance partials
// ---------------------------------------------------------------------------
__global__ void __launch_bounds__(256) moe_router(
    const float* __restrict__ x,
    const float* __restrict__ Wgp,     // [H, NEXP] prior gate
    const float* __restrict__ Wgq,     // [H, NEXP] posterior gate
    int* __restrict__ cnt, float* __restrict__ Psum, float* __restrict__ klsum,
    int* __restrict__ list, float* __restrict__ wlist) {
  __shared__ float sWp[H * NEXP];
  __shared__ float sWq[H * NEXP];
  __shared__ float sP[NEXP];
  __shared__ float sKL;

  const int tid = threadIdx.x;
  for (int i = tid; i < H * NEXP; i += 256) { sWp[i] = Wgp[i]; sWq[i] = Wgq[i]; }
  if (tid < NEXP) sP[tid] = 0.0f;
  if (tid == 0) sKL = 0.0f;
  __syncthreads();

  const int tok = blockIdx.x * 256 + tid;      // grid 8 x 256 == NTOK
  const int b = tok >> 10;
  const int l = tok & 1023;
  const int lf = (l + 2 > 1023) ? 1023 : (l + 2);   // lookforward shift, tail pad
  const float* xr = x + (size_t)tok * H;
  const float* xf = x + (size_t)((b << 10) + lf) * H;

  float p[NEXP], q[NEXP];
#pragma unroll
  for (int e = 0; e < NEXP; ++e) { p[e] = 0.0f; q[e] = 0.0f; }
  for (int j4 = 0; j4 < H / 4; ++j4) {
    const float4 xv = *(const float4*)&xr[j4 * 4];
    const float4 fv = *(const float4*)&xf[j4 * 4];
    const float xs[4] = {xv.x, xv.y, xv.z, xv.w};
    const float fs[4] = {fv.x, fv.y, fv.z, fv.w};
#pragma unroll
    for (int jj = 0; jj < 4; ++jj) {
      const int j = j4 * 4 + jj;
#pragma unroll
      for (int e = 0; e < NEXP; ++e) {
        p[e] = fmaf(xs[jj], sWp[j * NEXP + e], p[e]);
        q[e] = fmaf(fs[jj], sWq[j * NEXP + e], q[e]);
      }
    }
  }

  float mq = q[0];
#pragma unroll
  for (int e = 1; e < NEXP; ++e) mq = fmaxf(mq, q[e]);
  float eq[NEXP];
  float sq = 0.0f;
#pragma unroll
  for (int e = 0; e < NEXP; ++e) { eq[e] = __expf(q[e] - mq); sq += eq[e]; }
  const float invsq = 1.0f / sq;
  float pw[NEXP];
#pragma unroll
  for (int e = 0; e < NEXP; ++e) pw[e] = eq[e] * invsq;

  // top-2, jax tie-break = lowest index first (strict >)
  int i1 = 0; float v1 = pw[0];
#pragma unroll
  for (int e = 1; e < NEXP; ++e) if (pw[e] > v1) { v1 = pw[e]; i1 = e; }
  int i2 = -1; float v2 = -1.0f;
#pragma unroll
  for (int e = 0; e < NEXP; ++e) if (e != i1 && pw[e] > v2) { v2 = pw[e]; i2 = e; }
  const float wn = 1.0f / (v1 + v2);

  int p1 = atomicAdd(&cnt[i1], 1);
  list[i1 * NTOK + p1]  = tok;
  wlist[i1 * NTOK + p1] = v1 * wn;
  int p2 = atomicAdd(&cnt[i2], 1);
  list[i2 * NTOK + p2]  = tok;
  wlist[i2 * NTOK + p2] = v2 * wn;

#pragma unroll
  for (int e = 0; e < NEXP; ++e) atomicAdd(&sP[e], pw[e]);

  float mp = p[0];
#pragma unroll
  for (int e = 1; e < NEXP; ++e) mp = fmaxf(mp, p[e]);
  float sp = 0.0f;
#pragma unroll
  for (int e = 0; e < NEXP; ++e) sp += __expf(p[e] - mp);
  const float lse_p = __logf(sp);
  const float lse_q = __logf(sq);
  float kl = 0.0f;
#pragma unroll
  for (int e = 0; e < NEXP; ++e) {
    const float lq = (q[e] - mq) - lse_q;
    const float lp = (p[e] - mp) - lse_p;
    kl += pw[e] * (lq - lp);
  }
  atomicAdd(&sKL, kl);

  __syncthreads();
  if (tid < NEXP) atomicAdd(&Psum[tid], sP[tid]);
  if (tid == 0) atomicAdd(klsum, sKL);
}

// ---------------------------------------------------------------------------
// Kernel 3: expert FFN — fused bf16 WMMA (x@Wfc) -> relu^2 -> (@Wproj)
// Block = (token-tile of 64, expert e, chunk-group z). 512 threads = 16 waves.
// LDS (ush): XA 49152 | WfcB 49152 | WprojB 49152 | h1A 4096 | idx/w
// Async staging pipelined: WfcB(i+1) copies overlap GEMM2(i), WprojB(i+1)
// copies overlap GEMM1(i+1); buffers overwritten only after their readers'
// barrier, ASYNCcnt waited before each consuming phase.
// ---------------------------------------------------------------------------
#define SMEM_USH   (49152 + 49152 + 49152 + 4096)
#define SMEM_BYTES (SMEM_USH * 2 + TILE_M * 4 + TILE_M * 4)

__global__ void __launch_bounds__(NTHR) moe_expert_ffn(
    const float* __restrict__ x,
    const float* __restrict__ Wfc,
    const float* __restrict__ Wpj,
    const unsigned short* __restrict__ wfcp,   // prepped frags or null
    const unsigned short* __restrict__ wpjp,
    const int* __restrict__ cnt,
    const int* __restrict__ list,
    const float* __restrict__ wlist,
    float* __restrict__ dstbase,               // partials (G>1) or out (G==1)
    int nch_blk) {                             // chunks handled per block
  extern __shared__ unsigned short smem[];
  unsigned short* XA     = smem;               // [4][24][512]
  unsigned short* WfcB   = smem + 49152;       // [24][4][512]
  unsigned short* WprojB = smem + 98304;       // [2][48][512]
  unsigned short* h1A    = smem + 147456;      // [4][2][512]
  int*   sIdx = (int*)(smem + 151552);         // [64]
  float* sW   = (float*)(sIdx + TILE_M);       // [64]

  const int e    = blockIdx.y;
  const int tile = blockIdx.x;
  const int ne   = cnt[e];
  if (tile * TILE_M >= ne) return;

  const int tid  = threadIdx.x;
  const int wave = tid >> 5;
  const int lane = tid & 31;
  const bool prepped = (wfcp != nullptr);
  float* dst = dstbase + (size_t)blockIdx.z * NTOK * H;

  if (tid < TILE_M) {
    const int pos = tile * TILE_M + tid;
    const bool ok = pos < ne;
    sIdx[tid] = list[e * NTOK + (ok ? pos : 0)];
    sW[tid]   = ok ? wlist[e * NTOK + pos] : 0.0f;   // pad rows contribute 0
  }
  __syncthreads();

  // Gather + convert x tile into A-fragment-major LDS (float4 -> packed b64)
  for (int i = tid; i < TILE_M * H / 4; i += NTHR) {
    const int mm = i / (H / 4);
    const int c4 = (i - mm * (H / 4)) * 4;
    const float4 v = *(const float4*)&x[(size_t)sIdx[mm] * H + c4];
    const int mt = mm >> 4, m = mm & 15;
    const int t = c4 >> 5, kk = c4 & 31;
    int half, elem; a_map(kk, half, elem);
    unsigned short* d = XA + ((mt * KT1 + t) << 9) + ((half << 4) + m) * 16 + elem;
    u32x2 pk;
    pk.x = (unsigned)f2bf(v.x) | ((unsigned)f2bf(v.y) << 16);
    pk.y = (unsigned)f2bf(v.z) | ((unsigned)f2bf(v.w) << 16);
    *(u32x2*)d = pk;
  }

  v8f acc[12];
#pragma unroll
  for (int c = 0; c < 12; ++c)
#pragma unroll
    for (int r = 0; r < 8; ++r) acc[c][r] = 0.0f;

  const float* WfcE = Wfc + (size_t)e * H * FF;
  const float* WpjE = Wpj + (size_t)e * FF * H;
  const int mtile = wave & 3;     // token quarter this wave owns
  const int cgrp  = wave >> 2;    // col group: GEMM1 h1 ctile; GEMM2 12 ctiles
  const int ch0   = blockIdx.z * nch_blk;

  // --- shared GEMM phase bodies -------------------------------------------
  auto do_gemm1 = [&]() {
    v8f c1;
#pragma unroll
    for (int r = 0; r < 8; ++r) c1[r] = 0.0f;
    for (int t = 0; t < KT1; ++t) {
      Frag a, b;
      ldFrag(a, XA   + (((mtile * KT1 + t) << 9) + lane * 16));
      ldFrag(b, WfcB + (((t * CT1 + cgrp) << 9) + lane * 16));
      c1 = wmma_bf16(a, b, c1);
    }
#pragma unroll
    for (int r = 0; r < 8; ++r) {                 // relu^2 -> h1 A-frags
      float v = c1[r];
      v = (v > 0.0f) ? v * v : 0.0f;
      const int M    = r + ((lane >> 4) << 3);
      const int colc = (cgrp << 4) + (lane & 15);
      const int t  = colc >> 5, kk = colc & 31;
      int half, elem; a_map(kk, half, elem);
      h1A[((mtile * KT2 + t) << 9) + ((half << 4) + M) * 16 + elem] = f2bf(v);
    }
  };
  auto do_gemm2 = [&]() {
    Frag a0, a1;
    ldFrag(a0, h1A + (((mtile * KT2 + 0) << 9) + lane * 16));
    ldFrag(a1, h1A + (((mtile * KT2 + 1) << 9) + lane * 16));
#pragma unroll
    for (int cc2 = 0; cc2 < 12; ++cc2) {
      const int c = cgrp * 12 + cc2;
      Frag b0, b1;
      ldFrag(b0, WprojB + (((0 * CT2 + c) << 9) + lane * 16));
      ldFrag(b1, WprojB + (((1 * CT2 + c) << 9) + lane * 16));
      acc[cc2] = wmma_bf16(a0, b0, acc[cc2]);
      acc[cc2] = wmma_bf16(a1, b1, acc[cc2]);
    }
  };
  auto issue_fc = [&](int ch) {
    const unsigned short* g = wfcp + (size_t)(e * NCH + ch) * FRAG_CHUNK;
    for (int u = tid; u < FRAG_CHUNK / 8; u += NTHR) async16(g + u * 8, WfcB + u * 8);
  };
  auto issue_pj = [&](int ch) {
    const unsigned short* g = wpjp + (size_t)(e * NCH + ch) * FRAG_CHUNK;
    for (int u = tid; u < FRAG_CHUNK / 8; u += NTHR) async16(g + u * 8, WprojB + u * 8);
  };

  if (prepped) {
    // Prologue: stage chunk 0 fully.
    issue_fc(ch0);
    issue_pj(ch0);
    wait_async0();
    __syncthreads();
    for (int chi = 0; chi < nch_blk; ++chi) {
      const bool more = (chi + 1) < nch_blk;
      do_gemm1();                     // reads WfcB(i); WprojB(i) copies may be in flight
      wait_async0();
      __syncthreads();                // h1A ready; WfcB(i) consumed by all
      if (more) issue_fc(ch0 + chi + 1);   // overlaps GEMM2(i)
      do_gemm2();                     // reads WprojB(i), h1A
      wait_async0();                  // WfcB(i+1) landed (own waves)
      __syncthreads();                // WprojB(i) consumed by all
      if (more) issue_pj(ch0 + chi + 1);   // overlaps GEMM1(i+1)
    }
  } else {
    // Fallback: serial f32 -> bf16 conversion staging.
    for (int chi = 0; chi < nch_blk; ++chi) {
      const int ffc0 = (ch0 + chi) * CH;
      __syncthreads();
      for (int i = tid; i < H * CH; i += NTHR) {
        const int r  = i >> 6;
        const int cc = i & 63;
        const float v = WfcE[(size_t)r * FF + ffc0 + cc];
        const int t = r >> 5, kk = r & 31;
        int half, elem; b_map(kk, half, elem);
        const int c = cc >> 4, n = cc & 15;
        WfcB[((t * CT1 + c) << 9) + ((half << 4) + n) * 16 + elem] = f2bf(v);
      }
      for (int i = tid; i < CH * H; i += NTHR) {
        const int r  = i / H;
        const int cc = i - r * H;
        const float v = WpjE[(size_t)(ffc0 + r) * H + cc];
        const int t = r >> 5, kk = r & 31;
        int half, elem; b_map(kk, half, elem);
        const int c = cc >> 4, n = cc & 15;
        WprojB[((t * CT2 + c) << 9) + ((half << 4) + n) * 16 + elem] = f2bf(v);
      }
      __syncthreads();
      do_gemm1();
      __syncthreads();
      do_gemm2();
    }
  }

  // Weighted scatter-add. Per destination element exactly two commutative
  // f32 atomic adds (one per expert) -> bitwise deterministic.
#pragma unroll
  for (int cc2 = 0; cc2 < 12; ++cc2) {
    const int c = cgrp * 12 + cc2;
#pragma unroll
    for (int r = 0; r < 8; ++r) {
      const int M   = r + ((lane >> 4) << 3);
      const int mm  = (mtile << 4) + M;
      const int tok = sIdx[mm];
      const float w = sW[mm];
      const int col = (c << 4) + (lane & 15);
      atomicAdd(&dst[(size_t)tok * H + col], acc[cc2][r] * w);
    }
  }
}

// ---------------------------------------------------------------------------
// Kernel 4: deterministic fixed-order reduction of 4 partial buffers
// ---------------------------------------------------------------------------
__global__ void __launch_bounds__(256) moe_reduce(const float* __restrict__ partials,
                                                  float* __restrict__ out, int n4) {
  const int gid = blockIdx.x * blockDim.x + threadIdx.x;
  const float4* p0 = (const float4*)partials;
  const float4* p1 = p0 + n4;
  const float4* p2 = p1 + n4;
  const float4* p3 = p2 + n4;
  float4* o4 = (float4*)out;
  for (int i = gid; i < n4; i += gridDim.x * blockDim.x) {
    const float4 a = p0[i], b = p1[i], c = p2[i], d = p3[i];
    float4 r;
    r.x = (a.x + b.x) + (c.x + d.x);
    r.y = (a.y + b.y) + (c.y + d.y);
    r.z = (a.z + b.z) + (c.z + d.z);
    r.w = (a.w + b.w) + (c.w + d.w);
    o4[i] = r;
  }
}

// ---------------------------------------------------------------------------
// Kernel 5: aux loss scalar
// ---------------------------------------------------------------------------
__global__ void moe_finalize(const int* __restrict__ cnt,
                             const float* __restrict__ Psum,
                             const float* __restrict__ klsum,
                             float* __restrict__ out_aux) {
  if (threadIdx.x == 0 && blockIdx.x == 0) {
    float lb = 0.0f;
    for (int e = 0; e < NEXP; ++e) {
      const float f = (float)cnt[e] / ((float)NTOK * (float)TOPK);
      const float P = Psum[e] / (float)NTOK;
      lb += f * P;
    }
    lb *= (float)NEXP * 0.01f;               // LB_COEF
    const float kl = *klsum / (float)NTOK;   // KL_COEF = 1
    *out_aux = lb + kl;
  }
}

// ---------------------------------------------------------------------------
#define GSPLIT       4
#define WS_PART_OFF  262144
#define PART_BYTES   ((size_t)GSPLIT * NTOK * H * 4)
#define WS_PREP_OFF  (WS_PART_OFF + PART_BYTES)
#define PREP_USH     ((size_t)NEXP * NCH * FRAG_CHUNK)      // per weight array
#define WS_NEED_PART ((size_t)WS_PART_OFF + PART_BYTES)
#define WS_NEED_FULL (WS_PREP_OFF + 2 * PREP_USH * 2)

extern "C" void kernel_launch(void* const* d_in, const int* in_sizes, int n_in,
                              void* d_out, int out_size, void* d_ws, size_t ws_size,
                              hipStream_t stream) {
  const float* x   = (const float*)d_in[0];
  const float* Wgp = (const float*)d_in[1];
  const float* Wgq = (const float*)d_in[2];
  const float* Wfc = (const float*)d_in[3];
  const float* Wpj = (const float*)d_in[4];
  float* out = (float*)d_out;

  char* ws = (char*)d_ws;
  int*   cnt   = (int*)ws;                       // [8]
  float* Psum  = (float*)(ws + 32);              // [8]
  float* klsum = (float*)(ws + 64);              // [1]
  int*   list  = (int*)(ws + 128);               // [8][2048]
  float* wlist = (float*)(ws + 128 + NEXP * NTOK * 4);
  float* partials = (float*)(ws + WS_PART_OFF);  // [G][NTOK][H]

  const bool use_part = ws_size >= WS_NEED_PART;
  const bool use_prep = ws_size >= WS_NEED_FULL;
  unsigned short* wfcp = use_prep ? (unsigned short*)(ws + WS_PREP_OFF) : nullptr;
  unsigned short* wpjp = use_prep ? wfcp + PREP_USH : nullptr;

  const int G = use_part ? GSPLIT : 1;
  float* dstbase = use_part ? partials : out;

  (void)hipFuncSetAttribute((const void*)moe_expert_ffn,
                            hipFuncAttributeMaxDynamicSharedMemorySize,
                            SMEM_BYTES);

  const int nout = NTOK * H;   // 1572864 (+1 aux written by finalize)
  const int zero4 = (use_part ? G * nout : nout) / 4;
  moe_init<<<2048, 256, 0, stream>>>(dstbase, zero4, cnt, Psum, klsum);
  if (use_prep) {
    moe_prep<<<dim3(NCH, NEXP), 256, 0, stream>>>(Wfc, Wpj, wfcp, wpjp);
  }
  moe_router<<<NTOK / 256, 256, 0, stream>>>(x, Wgp, Wgq, cnt, Psum, klsum, list, wlist);
  moe_expert_ffn<<<dim3(NTOK / TILE_M, NEXP, G), NTHR, SMEM_BYTES, stream>>>(
      x, Wfc, Wpj, wfcp, wpjp, cnt, list, wlist, dstbase, NCH / G);
  if (use_part) {
    moe_reduce<<<1536, 256, 0, stream>>>(partials, out, nout / 4);
  }
  moe_finalize<<<1, 64, 0, stream>>>(cnt, Psum, klsum, out + nout);
}
